// Attention_84224308674843
// MI455X (gfx1250) — matrix operand order repaired and checked by reading, not compiled, not run on hardware
//
#include <hip/hip_runtime.h>
#include <hip/hip_bf16.h>

// ---------------------------------------------------------------------------
// CDNA5 (gfx1250) attention, bf16 WMMA + Tensor Data Mover staging.
// Pipeline:
//   1. kx  = k  @ w_kx            (WMMA GEMM, f32 in -> bf16 ws)
//   2. qx  = q  @ w_qx            (WMMA GEMM)
//   3. qw  = qx @ weight          (WMMA GEMM, A tile via TDM w/ LDS padding)
//   4. p   = exp(tanh(qw @ kx^T)) * mask ; rowsums
//        (fused kernel: TDM double-buffered kx tiles, native v_tanh_f32)
//   5. attn = p / rowsum          (scale kernel, in-place in d_out)
//   6. outh = attn @ kx           (batched WMMA GEMM)
//   7. out  = outh @ proj_w^T + b (WMMA GEMM, A tile via TDM, f32+bias out)
// ---------------------------------------------------------------------------

typedef __attribute__((ext_vector_type(16))) __bf16       bf16x16;
typedef __attribute__((ext_vector_type(8)))  float        v8f;
typedef __attribute__((ext_vector_type(4)))  unsigned int u32x4;
typedef __attribute__((ext_vector_type(8)))  int          i32x8;
typedef __attribute__((ext_vector_type(4)))  int          i32x4;

#define WMMA_BF16(a, b, c) \
  __builtin_amdgcn_wmma_f32_16x16x32_bf16(false, (a), false, (b), (short)0, (c), false, false)

#if __has_builtin(__builtin_amdgcn_tanhf)
#define TANHF(x) __builtin_amdgcn_tanhf(x)   // native v_tanh_f32 on gfx1250
#else
#define TANHF(x) tanhf(x)
#endif

union FragU {
  uint4 u[2];
  bf16x16 v;
};
union Pack4 { __bf16 h[4]; uint2 u; };
union Pack8 { __bf16 h[8]; uint4 u; };

// Load a v16bf fragment as two 16-byte chunks (ds_load_b128 / global_load_b128).
template <typename PT>
__device__ __forceinline__ bf16x16 load_frag2(const PT* p0, const PT* p1) {
  FragU f;
  f.u[0] = *reinterpret_cast<const uint4*>(p0);
  f.u[1] = *reinterpret_cast<const uint4*>(p1);
  return f.v;
}

// ---------------------------------------------------------------------------
// Tensor Data Mover: 2D bf16 tile (tile_d1 rows x tile_d0 elems, row stride
// stride_elems) from global -> LDS. Optional LDS padding (pad codes per D#:
// pad_interval: 0=2 DW,1=4,...,3=16 ; pad_amount: code = DWORDs-1).
// D# per CDNA5 ISA:
//   group0: count=1 | lds_addr[63:32] | global_addr[120:64] | type=2
//   group1: data_size=1 (2B) | pad | tensor_dim0/1 | tile_dim0/1 | dim0_stride
// ---------------------------------------------------------------------------
__device__ __forceinline__ void tdm_load_2d_bf16(unsigned lds_addr,
                                                 const void* gptr,
                                                 unsigned tile_d0, unsigned tile_d1,
                                                 unsigned long long stride_elems,
                                                 bool pad_en = false,
                                                 unsigned pad_interval_code = 0,
                                                 unsigned pad_amount_code = 0) {
  const unsigned long long ga = (unsigned long long)(size_t)gptr;
  u32x4 g0;
  g0[0] = 1u;                                                // count=1 (valid)
  g0[1] = lds_addr;                                          // LDS byte address
  g0[2] = (unsigned)(ga & 0xFFFFFFFFu);                      // global_addr lo
  g0[3] = (unsigned)((ga >> 32) & 0x1FFFFFFu) | (2u << 30);  // addr hi | type=2
  i32x8 g1;
  g1[0] = (int)((1u << 16) |                                 // data_size = 2 bytes
                (pad_en ? (1u << 20) : 0u) |
                (pad_interval_code << 22) | (pad_amount_code << 25));
  g1[1] = (int)((tile_d0 & 0xFFFFu) << 16);                  // tensor_dim0 lo16
  g1[2] = (int)((tile_d0 >> 16) | ((tile_d1 & 0xFFFFu) << 16));  // td0 hi | td1 lo
  g1[3] = (int)((tile_d1 >> 16) | (tile_d0 << 16));          // td1 hi | tile_dim0
  g1[4] = (int)(tile_d1 & 0xFFFFu);                          // tile_dim1 (tile_dim2=0)
  g1[5] = (int)(unsigned)(stride_elems & 0xFFFFFFFFu);       // dim0_stride lo32
  g1[6] = (int)((unsigned)(stride_elems >> 32) & 0xFFFFu);   // dim0_stride hi16
  g1[7] = 0;
  const i32x4 gz = {0, 0, 0, 0};
#if __clang_major__ >= 23
  const i32x8 gz8 = {0, 0, 0, 0, 0, 0, 0, 0};
  __builtin_amdgcn_tensor_load_to_lds(g0, g1, gz, gz, gz8, 0);
#else
  __builtin_amdgcn_tensor_load_to_lds(g0, g1, gz, gz, 0);
#endif
}

__device__ __forceinline__ unsigned lds_addr_of(const void* p) {
  return (unsigned)(unsigned long long)(size_t)p;  // LDS aperture: offset in [31:0]
}

// ---------------------------------------------------------------------------
// Generic tiled WMMA GEMM: C[M,N] = A[M,K] * W + (bias)
//   WTRANS=false: W is [K,N] row-major ; true: W is [N,K] row-major
//   ATDM  : A is bf16 and its 128x32 tile is staged by TENSOR_LOAD_TO_LDS
//           with pad_enable reproducing the 40-elem padded row stride.
// Block: 256 threads = 8 waves; 128x128 C tile. Wave (mg=w>>1, ng=w&1) owns a
// 32x64 sub-tile -> 8 WMMAs per K-step of 32. M,N divisible by 128; K by 32.
// All staging uses 16B global loads ; f32 A and [N,K] W use packed bf16 LDS stores, [K,N] W scalar ones.
// ---------------------------------------------------------------------------
template <typename AT, typename WT, bool WTRANS, bool OUTBF, bool BIAS, bool ATDM>
__global__ __launch_bounds__(256) void gemm_wmma_kernel(
    const AT* __restrict__ A, const WT* __restrict__ W,
    float* __restrict__ Cf, __bf16* __restrict__ Cb,
    const float* __restrict__ bias,
    int M, int N, int K,
    long long sA, long long sW, long long sC) {
  const int z = blockIdx.z;
  A += (long long)z * sA;
  W += (long long)z * sW;
  if (OUTBF) Cb += (long long)z * sC; else Cf += (long long)z * sC;

  const int tile_m = blockIdx.y * 128;
  const int tile_n = blockIdx.x * 128;
  const int tid  = threadIdx.x;   // 0..255
  const int wave = tid >> 5;      // 0..7
  const int lane = tid & 31;
  const int l16  = lane & 15;
  const int half = lane >> 4;
  const int mg   = wave >> 1;     // 0..3  (M group of 32 rows)
  const int ng   = wave & 1;      // 0..1  (N group of 64 cols)

  __shared__ __align__(16) __bf16 As[128][40];  // [m][k], row stride 80B
  __shared__ __align__(16) __bf16 Wt[128][40];  // [n][k]

  v8f acc[2][4] = {};

  for (int k0 = 0; k0 < K; k0 += 32) {
    // ---- stage A tile 128x32 ----
    if constexpr (ATDM) {
      // TDM: 128 rows x 32 bf16, LDS pad 4 DWORDs every 16 DWORDs -> stride 40
      if (wave == 0) {
        tdm_load_2d_bf16(lds_addr_of(&As[0][0]),
                         &A[(long long)tile_m * K + k0], 32, 128,
                         (unsigned long long)K,
                         /*pad_en=*/true, /*interval 16 DW=*/3, /*amount 4 DW=*/3);
      }
    } else {
      // f32 source: 1024 float4 loads, packed 4xbf16 (8B) LDS stores
      #pragma unroll
      for (int i = 0; i < 4; ++i) {
        const int i4 = tid + i * 256;        // 0..1023
        const int mm = i4 >> 3;              // 8 float4 per 32-elem row
        const int kc = (i4 & 7) * 4;
        const float4 v = *reinterpret_cast<const float4*>(
            &A[(long long)(tile_m + mm) * K + (k0 + kc)]);
        Pack4 p;
        p.h[0] = (__bf16)v.x; p.h[1] = (__bf16)v.y;
        p.h[2] = (__bf16)v.z; p.h[3] = (__bf16)v.w;
        *reinterpret_cast<uint2*>(&As[mm][kc]) = p.u;
      }
    }
    // ---- stage W tile as Wt[n][k] 128x32 ----
    if constexpr (WTRANS) {
      // W [N,K] row-major: contiguous k, same pattern as A
      #pragma unroll
      for (int i = 0; i < 4; ++i) {
        const int i4 = tid + i * 256;
        const int nn = i4 >> 3;
        const int kc = (i4 & 7) * 4;
        const float4 v = *reinterpret_cast<const float4*>(
            &W[(long long)(tile_n + nn) * K + (k0 + kc)]);
        Pack4 p;
        p.h[0] = (__bf16)v.x; p.h[1] = (__bf16)v.y;
        p.h[2] = (__bf16)v.z; p.h[3] = (__bf16)v.w;
        *reinterpret_cast<uint2*>(&Wt[nn][kc]) = p.u;
      }
    } else if constexpr (sizeof(WT) == 4) {
      // f32 W [K,N]: float4 along n, scatter-transpose into Wt[n][k]
      #pragma unroll
      for (int i = 0; i < 4; ++i) {
        const int i4 = tid + i * 256;        // 0..1023
        const int kk = i4 >> 5;              // 32 float4 per 128-elem row
        const int nc = (i4 & 31) * 4;
        const float4 v = *reinterpret_cast<const float4*>(
            &W[(long long)(k0 + kk) * N + (tile_n + nc)]);
        Wt[nc + 0][kk] = (__bf16)v.x;
        Wt[nc + 1][kk] = (__bf16)v.y;
        Wt[nc + 2][kk] = (__bf16)v.z;
        Wt[nc + 3][kk] = (__bf16)v.w;
      }
    } else {
      // bf16 W [K,N]: uint4 (8 elems) along n, scatter-transpose
      #pragma unroll
      for (int i = 0; i < 2; ++i) {
        const int i8 = tid + i * 256;        // 0..511
        const int kk = i8 >> 4;              // 16 uint4 per 128-elem row
        const int nc = (i8 & 15) * 8;
        Pack8 p;
        p.u = *reinterpret_cast<const uint4*>(
            &W[(long long)(k0 + kk) * N + (tile_n + nc)]);
        #pragma unroll
        for (int j = 0; j < 8; ++j) Wt[nc + j][kk] = p.h[j];
      }
    }
    if (k0 + 32 < K) {  // hint next A rows into cache (global_prefetch_b8)
      __builtin_prefetch(&A[(long long)(tile_m + (tid >> 1)) * K + (k0 + 32)], 0, 1);
    }
    if constexpr (ATDM) {
      if (wave == 0) __builtin_amdgcn_s_wait_tensorcnt(0);
    }
    __syncthreads();

    // ---- preload all fragments (named, kept live), then 8 WMMAs ----
    const __bf16* ar0 = &As[mg * 32 + l16][0];
    const __bf16* ar1 = &As[mg * 32 + 16 + l16][0];
    const bf16x16 af0 = load_frag2(ar0 + half * 16, ar0 + half * 16 + 8);
    const bf16x16 af1 = load_frag2(ar1 + half * 16, ar1 + half * 16 + 8);
    const __bf16* br0 = &Wt[ng * 64 + 0 * 16 + l16][half * 16];
    const __bf16* br1 = &Wt[ng * 64 + 1 * 16 + l16][half * 16];
    const __bf16* br2 = &Wt[ng * 64 + 2 * 16 + l16][half * 16];
    const __bf16* br3 = &Wt[ng * 64 + 3 * 16 + l16][half * 16];
    const bf16x16 bf0 = load_frag2(br0, br0 + 8);
    const bf16x16 bf1 = load_frag2(br1, br1 + 8);
    const bf16x16 bf2 = load_frag2(br2, br2 + 8);
    const bf16x16 bf3 = load_frag2(br3, br3 + 8);
    acc[0][0] = WMMA_BF16(af0, bf0, acc[0][0]);
    acc[1][0] = WMMA_BF16(af1, bf0, acc[1][0]);
    acc[0][1] = WMMA_BF16(af0, bf1, acc[0][1]);
    acc[1][1] = WMMA_BF16(af1, bf1, acc[1][1]);
    acc[0][2] = WMMA_BF16(af0, bf2, acc[0][2]);
    acc[1][2] = WMMA_BF16(af1, bf2, acc[1][2]);
    acc[0][3] = WMMA_BF16(af0, bf3, acc[0][3]);
    acc[1][3] = WMMA_BF16(af1, bf3, acc[1][3]);
    __syncthreads();
  }

  // ---- store: C layout — lane column N=l16, VGPR r -> row r + half*8 ----
  #pragma unroll
  for (int mi = 0; mi < 2; ++mi) {
    #pragma unroll
    for (int j = 0; j < 4; ++j) {
      const int n = tile_n + ng * 64 + j * 16 + l16;
      #pragma unroll
      for (int r = 0; r < 8; ++r) {
        const int m = tile_m + mg * 32 + mi * 16 + half * 8 + r;
        float v = acc[mi][j][r];
        if (BIAS) v += bias[n];
        if (OUTBF) Cb[(long long)m * N + n] = (__bf16)v;
        else       Cf[(long long)m * N + n] = v;
      }
    }
  }
}

// ---------------------------------------------------------------------------
// Fused score kernel. Block = 256 thr (8 waves) handles (batch b, 16 q rows).
//  - qw row block (16x1024 bf16) TDM'd to LDS once.
//  - keys processed in 8 outer steps of 512; per h-step the 512x32 bf16 kx
//    tile is TDM'd into a double-buffered LDS tile (wave 0 issues, overlap:
//    issue next -> compute current -> s_wait_tensorcnt 0 -> barrier).
//  - each wave keeps 4 key-tile accumulators: 1 A-frag feeds 4 WMMAs/h-step.
//  - epilogue: p = exp(tanh(S))*mask (native v_tanh_f32/v_exp_f32),
//    store unnormalized p, LDS rowsums via ds_add_f32.
// ---------------------------------------------------------------------------
__global__ __launch_bounds__(256) void attn_score_kernel(
    const __bf16* __restrict__ qw, const __bf16* __restrict__ kx,
    const int* __restrict__ memory_len,
    float* __restrict__ attn,        // unnormalized p, [B*QL, KL]
    float* __restrict__ rsum_recip,  // [B*QL]
    int B_, int QL_, int KL_, int H_) {
  const int b  = blockIdx.y;
  const int q0 = blockIdx.x * 16;
  const int tid  = threadIdx.x;
  const int wave = tid >> 5;
  const int lane = tid & 31;
  const int l16  = lane & 15;
  const int half = lane >> 4;
  const int ml   = memory_len[b];

  __shared__ __align__(16) __bf16 qw_s[16][1024];    // 32 KB
  __shared__ __align__(16) __bf16 kx_s[2][512][32];  // 64 KB double buffer
  __shared__ float rowsum[16];
  if (tid < 16) rowsum[tid] = 0.0f;

  const __bf16* qw_base = qw + ((long long)b * QL_ + q0) * H_;
  const __bf16* kx_base = kx + (long long)b * KL_ * H_;
  float* attn_base = attn + ((long long)b * QL_ + q0) * KL_;

  const int nko = KL_ / 512;        // 8 outer key steps
  const int nhs = H_ / 32;          // 32 h-steps
  const int nsteps = nko * nhs;

  if (wave == 0) {
    tdm_load_2d_bf16(lds_addr_of(&qw_s[0][0]), qw_base, (unsigned)H_, 16,
                     (unsigned long long)H_);
    tdm_load_2d_bf16(lds_addr_of(&kx_s[0][0][0]), kx_base, 32, 512,
                     (unsigned long long)H_);
    __builtin_amdgcn_s_wait_tensorcnt(0);
  }
  __syncthreads();

  for (int ko = 0; ko < nko; ++ko) {
    v8f acc[4] = {};
    for (int hs = 0; hs < nhs; ++hs) {
      const int step = ko * nhs + hs;
      const int cur  = step & 1;
      if (wave == 0 && step + 1 < nsteps) {  // issue next tile into other buffer
        const int ns  = step + 1;
        const int nk  = ns >> 5, nh = ns & 31;
        tdm_load_2d_bf16(lds_addr_of(&kx_s[cur ^ 1][0][0]),
                         kx_base + ((long long)nk * 512) * H_ + nh * 32,
                         32, 512, (unsigned long long)H_);
      }
      // preload 1 A-frag + 4 named B-frags (kept live), then 4 WMMAs
      const __bf16* arow = &qw_s[l16][hs * 32];
      const bf16x16 afrag = load_frag2(arow + half * 16, arow + half * 16 + 8);
      const __bf16* b0 = &kx_s[cur][(wave * 4 + 0) * 16 + l16][half * 16];
      const __bf16* b1 = &kx_s[cur][(wave * 4 + 1) * 16 + l16][half * 16];
      const __bf16* b2 = &kx_s[cur][(wave * 4 + 2) * 16 + l16][half * 16];
      const __bf16* b3 = &kx_s[cur][(wave * 4 + 3) * 16 + l16][half * 16];
      const bf16x16 bf0 = load_frag2(b0, b0 + 8);
      const bf16x16 bf1 = load_frag2(b1, b1 + 8);
      const bf16x16 bf2 = load_frag2(b2, b2 + 8);
      const bf16x16 bf3 = load_frag2(b3, b3 + 8);
      acc[0] = WMMA_BF16(afrag, bf0, acc[0]);
      acc[1] = WMMA_BF16(afrag, bf1, acc[1]);
      acc[2] = WMMA_BF16(afrag, bf2, acc[2]);
      acc[3] = WMMA_BF16(afrag, bf3, acc[3]);
      if (wave == 0) __builtin_amdgcn_s_wait_tensorcnt(0);
      __syncthreads();
    }
    // ---- epilogue for the 4 key tiles of this outer step ----
    #pragma unroll
    for (int t = 0; t < 4; ++t) {
      const int col = (ko * 32 + wave * 4 + t) * 16 + l16;
      const float maskf = (col < ml) ? 1.0f : 0.0f;
      #pragma unroll
      for (int r = 0; r < 8; ++r) {
        const float p = __expf(TANHF(acc[t][r])) * maskf;
        attn_base[(long long)(half * 8 + r) * KL_ + col] = p;
        float v = p;  // sum across the 16-lane column group
        #pragma unroll
        for (int off = 8; off >= 1; off >>= 1) v += __shfl_xor(v, off, 32);
        if (l16 == 0) atomicAdd(&rowsum[half * 8 + r], v);  // ds_add_f32
      }
    }
  }
  __syncthreads();
  if (tid < 16) rsum_recip[(long long)b * QL_ + q0 + tid] = 1.0f / rowsum[tid];
}

// Normalize attn in place: attn[row, :] *= rsum_recip[row].  float4 chunks.
// kl_shift = log2(KL) so the row index is a shift, not a 64-bit divide.
__global__ __launch_bounds__(256) void attn_scale_kernel(
    float* __restrict__ attn, const float* __restrict__ rsum_recip,
    long long total, int kl_shift) {
  long long idx = ((long long)blockIdx.x * blockDim.x + threadIdx.x) * 4;
  if (idx >= total) return;
  const float s = rsum_recip[idx >> kl_shift];  // KL%4==0: never crosses rows
  float4 v = *reinterpret_cast<float4*>(attn + idx);
  v.x *= s; v.y *= s; v.z *= s; v.w *= s;
  *reinterpret_cast<float4*>(attn + idx) = v;
}

// ---------------------------------------------------------------------------
extern "C" void kernel_launch(void* const* d_in, const int* in_sizes, int n_in,
                              void* d_out, int out_size, void* d_ws, size_t ws_size,
                              hipStream_t stream) {
  constexpr int B = 16, KL = 4096, QL = 1024, E = 1024, H = 1024;

  const float* k_in   = (const float*)d_in[0];
  const float* q_in   = (const float*)d_in[1];
  const int*   memlen = (const int*)d_in[2];
  const float* w_kx   = (const float*)d_in[3];
  const float* w_qx   = (const float*)d_in[4];
  const float* weight = (const float*)d_in[5];
  const float* proj_w = (const float*)d_in[6];
  const float* proj_b = (const float*)d_in[7];

  float* out_f  = (float*)d_out;                  // [B*QL, E]
  float* attn_f = out_f + (long long)B * QL * E;  // [B*QL, KL]

  // ---- workspace carve (≈235 MB) ----
  char* w = (char*)d_ws;
  __bf16* kx_bf   = (__bf16*)w;  w += (size_t)B * KL * H * 2;
  __bf16* qx_bf   = (__bf16*)w;  w += (size_t)B * QL * H * 2;
  __bf16* qw_bf   = (__bf16*)w;  w += (size_t)B * QL * H * 2;
  __bf16* outh_bf = (__bf16*)w;  w += (size_t)B * QL * H * 2;
  float*  rsum    = (float*)w;   w += (size_t)B * QL * 4;

  // 1. kx = k @ w_kx   [65536,1024] x [1024,1024] -> bf16
  gemm_wmma_kernel<float, float, false, true, false, false>
      <<<dim3(H / 128, (B * KL) / 128, 1), 256, 0, stream>>>(
          k_in, w_kx, nullptr, kx_bf, nullptr, B * KL, H, E, 0, 0, 0);

  // 2. qx = q @ w_qx   [16384,1024] -> bf16
  gemm_wmma_kernel<float, float, false, true, false, false>
      <<<dim3(H / 128, (B * QL) / 128, 1), 256, 0, stream>>>(
          q_in, w_qx, nullptr, qx_bf, nullptr, B * QL, H, E, 0, 0, 0);

  // 3. qw = qx @ weight  (bf16 A via TDM, f32 W) -> bf16
  gemm_wmma_kernel<__bf16, float, false, true, false, true>
      <<<dim3(H / 128, (B * QL) / 128, 1), 256, 0, stream>>>(
          qx_bf, weight, nullptr, qw_bf, nullptr, B * QL, H, H, 0, 0, 0);

  // 4. p = exp(tanh(qw @ kx^T)) * mask ; rowsums   (TDM-staged)
  attn_score_kernel<<<dim3(QL / 16, B, 1), 256, 0, stream>>>(
      qw_bf, kx_bf, memlen, attn_f, rsum, B, QL, KL, H);

  // 5. attn = p / rowsum (in place; this is the 2nd returned output)
  {
    const long long total = (long long)B * QL * KL;
    const int blocks = (int)((total / 4 + 255) / 256);
    attn_scale_kernel<<<blocks, 256, 0, stream>>>(attn_f, rsum, total, 12);
  }

  // 6. outh = attn @ kx   batched: per-batch [1024,4096] x [4096,1024] -> bf16
  gemm_wmma_kernel<float, __bf16, false, true, false, false>
      <<<dim3(H / 128, QL / 128, B), 256, 0, stream>>>(
          attn_f, kx_bf, nullptr, outh_bf, nullptr, QL, H, KL,
          (long long)QL * KL, (long long)KL * H, (long long)QL * H);

  // 7. out = outh @ proj_w^T + proj_b   (bf16 A via TDM; proj_w [E,H] -> WTRANS)
  gemm_wmma_kernel<__bf16, float, true, false, true, true>
      <<<dim3(E / 128, (B * QL) / 128, 1), 256, 0, stream>>>(
          outh_bf, proj_w, out_f, nullptr, proj_b, B * QL, E, H, 0, 0, 0);
}
